// NearestNeighbor_20358144983611
// MI455X (gfx1250) — matrix operand order repaired and checked by reading, not compiled, not run on hardware
//
#include <hip/hip_runtime.h>
#include <hip/hip_bf16.h>
#include <math.h>

// ---------------------------------------------------------------------------
// Nearest-neighbor (L2) search, MI455X / gfx1250.
//
//   d2(b,n) = ||x_b||^2 + ||s_n||^2 - 2 * dot(x_b, s_n)
//
// GEMM core: 2-term bf16 split of fp32 (hi + lo), 3x V_WMMA_F32_16X16X32_BF16
// cross terms, fp32 accumulate  ->  ~fp32 accuracy at the bf16 matrix rate.
// Fused row-min/argmin (never materializes the 128x50000 distance matrix),
// ||s||^2 folded into the streaming pass, ||x||^2 deferred to the finalize.
// Cross-workgroup argmin via sortable-key u64 global_atomic_min.
// ---------------------------------------------------------------------------

typedef __attribute__((ext_vector_type(4)))  __bf16 v4bf;
typedef __attribute__((ext_vector_type(8)))  __bf16 v8bf;
typedef __attribute__((ext_vector_type(16))) __bf16 v16bf;
typedef __attribute__((ext_vector_type(8)))  float  v8f;

#define DIMD   3072            // C*H*W
#define BATCH  128
#define NCLS   10
#define NTILE  80              // samples per workgroup (5 WMMA N-tiles of 16)
#define KC     128             // K chunk staged in LDS
#define PITCH  136             // LDS row pitch (bf16 elems): 272B -> 16B aligned,
                               // rows offset by 4 banks -> conflict-free B reads

// ---------------------------------------------------------------------------
// Kernel 1: split X (128 x 3072 fp32) into bf16 hi/lo planes (L2-resident),
// compute ||x_b||^2, and reset the per-row argmin slots.
// ---------------------------------------------------------------------------
__global__ __launch_bounds__(256) void nn_prep(
    const float* __restrict__ x,
    __bf16* __restrict__ xhi, __bf16* __restrict__ xlo,
    float* __restrict__ x2, unsigned long long* __restrict__ best)
{
    __shared__ float red[256];
    const int b = blockIdx.x, t = threadIdx.x;
    const float* row = x + (size_t)b * DIMD;
    float part = 0.0f;
#pragma unroll
    for (int j = 0; j < 3; ++j) {
        const int f = (t + 256 * j) * 4;
        const float4 v = *(const float4*)(row + f);
        const float vv[4] = {v.x, v.y, v.z, v.w};
        v4bf h, l;
#pragma unroll
        for (int e = 0; e < 4; ++e) {
            const __bf16 hb = (__bf16)vv[e];
            h[e] = hb;
            l[e] = (__bf16)(vv[e] - (float)hb);
            part += vv[e] * vv[e];
        }
        *(v4bf*)(xhi + (size_t)b * DIMD + f) = h;
        *(v4bf*)(xlo + (size_t)b * DIMD + f) = l;
    }
    red[t] = part;
    __syncthreads();
    for (int s = 128; s > 0; s >>= 1) {
        if (t < s) red[t] += red[t + s];
        __syncthreads();
    }
    if (t == 0) { x2[b] = red[0]; best[b] = ~0ull; }
}

// ---------------------------------------------------------------------------
// Kernel 2: main fused GEMM + argmin. 625 workgroups x 256 threads (8 waves).
// Each workgroup: 80 samples x all 128 batch rows, K streamed in 128-chunks.
// Wave w owns batch rows [16w, 16w+16); 5 accumulator tiles of 16x16 f32.
// ---------------------------------------------------------------------------
__global__ __launch_bounds__(256) void nn_main(
    const float* __restrict__ S,
    const __bf16* __restrict__ xhi, const __bf16* __restrict__ xlo,
    unsigned long long* __restrict__ best)
{
    __shared__ __align__(16) __bf16 s_hi[NTILE * PITCH];
    __shared__ __align__(16) __bf16 s_lo[NTILE * PITCH];
    __shared__ float s_s2[NTILE];

    const int t    = threadIdx.x;
    const int lane = t & 31;
    const int w    = t >> 5;
    const int lrow = lane & 15;      // row-within-tile for A/B/C fragments
    const int ksel = lane >> 4;      // half-wave select (ISA bf16 VGPR layout)
    const int n0   = blockIdx.x * NTILE;

    if (t < NTILE) s_s2[t] = 0.0f;
    __syncthreads();

    v8f acc[5];
#pragma unroll
    for (int nt = 0; nt < 5; ++nt) acc[nt] = (v8f)0.0f;

    const __bf16* xh = xhi + (size_t)(w * 16 + lrow) * DIMD;
    const __bf16* xl = xlo + (size_t)(w * 16 + lrow) * DIMD;

    for (int c = 0; c < DIMD / KC; ++c) {
        const int kbase = c * KC;

        // ---- stream S chunk: global fp32 -> (bf16 hi, bf16 lo) in LDS ------
        // 80 samples x 128 K = 2560 float4 across 256 threads (10 each).
#pragma unroll
        for (int j = 0; j < 10; ++j) {
            const int f   = t + 256 * j;
            const int smp = f >> 5;          // 0..79
            const int k4  = (f & 31) * 4;    // 0..124
            const float* gp = S + (size_t)(n0 + smp) * DIMD + kbase + k4;
            const float4 v = *(const float4*)gp;
            if (c + 1 < DIMD / KC)
                __builtin_prefetch(gp + KC, 0, 1);   // global_prefetch_b8: next chunk
            const float vv[4] = {v.x, v.y, v.z, v.w};
            v4bf h, l;
            float dp = 0.0f;
#pragma unroll
            for (int e = 0; e < 4; ++e) {
                const __bf16 hb = (__bf16)vv[e];
                h[e] = hb;
                l[e] = (__bf16)(vv[e] - (float)hb);
                dp += vv[e] * vv[e];
            }
            *(v4bf*)&s_hi[smp * PITCH + k4] = h;
            *(v4bf*)&s_lo[smp * PITCH + k4] = l;
            atomicAdd(&s_s2[smp], dp);               // ds_add_f32: ||s||^2 on the fly
        }
        __syncthreads();

        // ---- 3-term bf16 WMMA over the chunk: 4 K-steps x 5 tiles x 3 ------
#pragma unroll
        for (int ks = 0; ks < KC / 32; ++ks) {
            // A fragment (16x32 bf16, ISA layout): lane half selects K runs
            //   lanes 0-15: K {0..7, 16..23}; lanes 16-31: K {8..15, 24..31}
            const int ka = kbase + ks * 32 + ksel * 8;
            const v8bf a0h = *(const v8bf*)(xh + ka);
            const v8bf a1h = *(const v8bf*)(xh + ka + 16);
            const v8bf a0l = *(const v8bf*)(xl + ka);
            const v8bf a1l = *(const v8bf*)(xl + ka + 16);
            const v16bf Ah = __builtin_shufflevector(a0h, a1h,
                0,1,2,3,4,5,6,7,8,9,10,11,12,13,14,15);
            const v16bf Al = __builtin_shufflevector(a0l, a1l,
                0,1,2,3,4,5,6,7,8,9,10,11,12,13,14,15);

            const int kb = ks * 32 + ksel * 8;       // B: lanes 0-15 K {0..7,16..23}, 16-31 K {8..15,24..31}
#pragma unroll
            for (int nt = 0; nt < 5; ++nt) {
                const __bf16* bh = &s_hi[(nt * 16 + lrow) * PITCH + kb];
                const __bf16* bl = &s_lo[(nt * 16 + lrow) * PITCH + kb];
                const v16bf Bh = __builtin_shufflevector(
                    *(const v8bf*)bh, *(const v8bf*)(bh + 16),
                    0,1,2,3,4,5,6,7,8,9,10,11,12,13,14,15);
                const v16bf Bl = __builtin_shufflevector(
                    *(const v8bf*)bl, *(const v8bf*)(bl + 16),
                    0,1,2,3,4,5,6,7,8,9,10,11,12,13,14,15);
                acc[nt] = __builtin_amdgcn_wmma_f32_16x16x32_bf16(
                    false, Ah, false, Bh, (short)0, acc[nt], false, false);
                acc[nt] = __builtin_amdgcn_wmma_f32_16x16x32_bf16(
                    false, Ah, false, Bl, (short)0, acc[nt], false, false);
                acc[nt] = __builtin_amdgcn_wmma_f32_16x16x32_bf16(
                    false, Al, false, Bh, (short)0, acc[nt], false, false);
            }
        }
        __syncthreads();
    }

    // ---- fused min/argmin epilogue ----------------------------------------
    // C layout: VGPR r, lanes 0-15 -> (M=r, N=lane); lanes 16-31 -> (M=8+r).
    // key = ||s||^2 - 2*dot  (||x||^2 is row-constant -> argmin-invariant).
#pragma unroll
    for (int r = 0; r < 8; ++r) {
        unsigned long long m = ~0ull;
#pragma unroll
        for (int nt = 0; nt < 5; ++nt) {
            const int nl = nt * 16 + lrow;
            const float v = s_s2[nl] - 2.0f * acc[nt][r];
            unsigned int bits = __float_as_uint(v);
            bits = (bits & 0x80000000u) ? ~bits : (bits | 0x80000000u); // order-preserving
            const unsigned long long pk =
                ((unsigned long long)bits << 32) | (unsigned int)(n0 + nl);
            if (pk < m) m = pk;
        }
#pragma unroll
        for (int off = 1; off < 16; off <<= 1) {     // reduce across the 16 N lanes
            const unsigned long long o = __shfl_xor(m, off, 32);
            if (o < m) m = o;
        }
        if (lrow == 0)
            atomicMin(&best[w * 16 + r + 8 * ksel], m);  // global_atomic_min_u64
    }
}

// ---------------------------------------------------------------------------
// Kernel 3: decode argmin, emit one-hot pred, gather winning image, l2s.
// ---------------------------------------------------------------------------
__global__ __launch_bounds__(256) void nn_final(
    const float* __restrict__ S, const long long* __restrict__ cls,
    const float* __restrict__ x2, const unsigned long long* __restrict__ best,
    float* __restrict__ pred, float* __restrict__ imgs, float* __restrict__ l2s)
{
    const int b = blockIdx.x, t = threadIdx.x;
    const unsigned long long k = best[b];
    const unsigned int idx = (unsigned int)(k & 0xFFFFFFFFu);
    unsigned int sb = (unsigned int)(k >> 32);
    sb = (sb & 0x80000000u) ? (sb ^ 0x80000000u) : ~sb;  // undo sortable map
    const float v = __uint_as_float(sb);                 // min(||s||^2 - 2 dot)

    if (t < NCLS)
        pred[b * NCLS + t] = (cls[idx] == (long long)t) ? 1.0f : 0.0f;
    if (t == 0)
        l2s[b] = sqrtf(fmaxf(x2[b] + v, 0.0f));

    const float* src = S + (size_t)idx * DIMD;
    float* dst = imgs + (size_t)b * DIMD;
#pragma unroll
    for (int j = 0; j < 3; ++j) {
        const int f = (t + 256 * j) * 4;
        *(float4*)(dst + f) = *(const float4*)(src + f);
    }
}

// ---------------------------------------------------------------------------
extern "C" void kernel_launch(void* const* d_in, const int* in_sizes, int n_in,
                              void* d_out, int out_size, void* d_ws, size_t ws_size,
                              hipStream_t stream)
{
    (void)n_in; (void)out_size; (void)ws_size;
    const float*     x   = (const float*)d_in[0];       // (128,3,32,32) fp32
    const float*     S   = (const float*)d_in[1];       // (50000,3,32,32) fp32
    const long long* cls = (const long long*)d_in[2];   // (50000,) int64
    const int N = in_sizes[2];                          // 50000

    // workspace layout (~1.5 MB): X hi plane | X lo plane | x2 | best keys
    char* ws = (char*)d_ws;
    __bf16* xhi = (__bf16*)ws;
    __bf16* xlo = (__bf16*)(ws + (size_t)BATCH * DIMD * 2);
    float*  x2  = (float*) (ws + (size_t)BATCH * DIMD * 4);
    unsigned long long* best =
        (unsigned long long*)(ws + (size_t)BATCH * DIMD * 4 + 512);

    float* pred = (float*)d_out;                 // (128,10)
    float* imgs = pred + BATCH * NCLS;           // (128,3072)
    float* l2s  = imgs + (size_t)BATCH * DIMD;   // (128,)

    nn_prep <<<dim3(BATCH),      dim3(256), 0, stream>>>(x, xhi, xlo, x2, best);
    nn_main <<<dim3(N / NTILE),  dim3(256), 0, stream>>>(S, xhi, xlo, best);
    nn_final<<<dim3(BATCH),      dim3(256), 0, stream>>>(S, cls, x2, best,
                                                         pred, imgs, l2s);
}
